// TransformerBlock_46016279609781
// MI455X (gfx1250) — compile-verified, matrix operand order checked by reading, not run on hardware
//
#include <hip/hip_runtime.h>
#include <hip/hip_bf16.h>

// ---------------- problem constants ----------------
#define BDIM   4
#define SEQ    2048
#define DMODEL 1024
#define NHEAD  16
#define HDIM   64
#define DFFN   4096
#define MTOK   (BDIM * SEQ)          // 8192 tokens

static constexpr float RES_C    = 0.28867513459481287f;  // 1/sqrt(12)
static constexpr float SCALE_C  = 0.03125f;              // 1/sqrt(1024)
static constexpr float LOG2E    = 1.4426950408889634f;
static constexpr float SCL2_C   = SCALE_C * LOG2E;       // score scale, log2 domain
static constexpr float EPS_C    = 1e-5f;

// ---------------- vector types for WMMA ----------------
typedef __attribute__((ext_vector_type(16))) __bf16        v16bf;
typedef __attribute__((ext_vector_type(8)))  float         v8f;
typedef __attribute__((ext_vector_type(4)))  unsigned short v4u;
typedef __attribute__((ext_vector_type(8)))  unsigned short v8u;
typedef __attribute__((ext_vector_type(16))) unsigned short v16u;

typedef unsigned short us;

__device__ __forceinline__ us f2bf(float f) {
    unsigned u = __float_as_uint(f);
    u = u + 0x7FFFu + ((u >> 16) & 1u);      // round-to-nearest-even
    return (us)(u >> 16);
}

__device__ __forceinline__ v8f wmma_bf16(v16bf a, v16bf b, v8f c) {
    return __builtin_amdgcn_wmma_f32_16x16x32_bf16(
        false, a, false, b, (short)0, c, false, false);
}

// CDNA5 async global->LDS copy (ASYNCcnt-tracked), 16 bytes per lane.
__device__ __forceinline__ void async_ld_b128(void* lds, const void* gptr) {
    asm volatile("global_load_async_to_lds_b128 %0, %1, off"
                 :: "v"((unsigned)(uintptr_t)lds), "v"(gptr)
                 : "memory");
}
__device__ __forceinline__ void wait_async0() {
    asm volatile("s_wait_asynccnt 0x0" ::: "memory");
}

// swap values between the two 16-lane halves of a wave32 (xor-16), pure VALU
__device__ __forceinline__ float xhalf_swap(float x) {
    const int i = __builtin_amdgcn_permlanex16(
        __float_as_int(x), __float_as_int(x),
        0x76543210, 0xfedcba98, false, false);
    return __int_as_float(i);
}

// Load a 16-bit A/B fragment for one 16x32 (rows x K) tile.
// base points at (row r, k=0); hi = lane>>4:
//   chunk0 = K[hi*8 .. hi*8+7], chunk1 = K[16+hi*8 .. 23+hi*8]
__device__ __forceinline__ v16bf frag_ld(const us* base, int hi) {
    const v8u lo = *(const v8u*)(base + hi * 8);
    const v8u hv = *(const v8u*)(base + 16 + hi * 8);
    v16u c;
#pragma unroll
    for (int i = 0; i < 8; ++i) { c[i] = lo[i]; c[i + 8] = hv[i]; }
    return __builtin_bit_cast(v16bf, c);
}

// Build a fragment directly from two in-register 8-element chunks.
__device__ __forceinline__ v16bf frag_pack(v8u lo, v8u hv) {
    v16u c;
#pragma unroll
    for (int i = 0; i < 8; ++i) { c[i] = lo[i]; c[i + 8] = hv[i]; }
    return __builtin_bit_cast(v16bf, c);
}

// -------- weight convert+transpose (LDS tiled): W[K,N] f32 -> Wt[N,K] bf16 --
__global__ void k_transpose_bf16(const float* __restrict__ W,
                                 us* __restrict__ Wt, int K, int N) {
    __shared__ float T[32][33];
    const int tx = threadIdx.x & 31, ty = threadIdx.x >> 5;  // 32 x 8
    const int k0 = blockIdx.y * 32, n0 = blockIdx.x * 32;
#pragma unroll
    for (int i = 0; i < 4; ++i) {
        const int kr = ty + i * 8;
        T[kr][tx] = W[(size_t)(k0 + kr) * N + n0 + tx];
    }
    __syncthreads();
#pragma unroll
    for (int i = 0; i < 4; ++i) {
        const int nr = ty + i * 8;
        Wt[(size_t)(n0 + nr) * K + k0 + tx] = f2bf(T[tx][nr]);
    }
}

// ---------------- LayerNorm: f32 [row, D] -> bf16 ----------------
__global__ void k_layernorm(const float* __restrict__ x,
                            const float* __restrict__ g,
                            const float* __restrict__ be,
                            us* __restrict__ y) {
    __shared__ float red[256];
    const int row = blockIdx.x, tid = threadIdx.x;
    const float* xr = x + (size_t)row * DMODEL;
    float v[4];
#pragma unroll
    for (int i = 0; i < 4; ++i) v[i] = xr[tid + i * 256];
    red[tid] = v[0] + v[1] + v[2] + v[3]; __syncthreads();
#pragma unroll
    for (int o = 128; o > 0; o >>= 1) { if (tid < o) red[tid] += red[tid + o]; __syncthreads(); }
    const float mean = red[0] * (1.0f / DMODEL);
    __syncthreads();
    float s2 = 0.f;
#pragma unroll
    for (int i = 0; i < 4; ++i) { float d = v[i] - mean; s2 += d * d; }
    red[tid] = s2; __syncthreads();
#pragma unroll
    for (int o = 128; o > 0; o >>= 1) { if (tid < o) red[tid] += red[tid + o]; __syncthreads(); }
    const float rstd = rsqrtf(red[0] * (1.0f / DMODEL) + EPS_C);
#pragma unroll
    for (int i = 0; i < 4; ++i) {
        const int col = tid + i * 256;
        y[(size_t)row * DMODEL + col] = f2bf((v[i] - mean) * rstd * g[col] + be[col]);
    }
}

// -- WMMA GEMM, async-LDS double buffered: C[M,N] = A[M,K] * Bt[N,K]^T ------
// EPI: 0 = bias -> bf16 ; 1 = relu(bias) -> bf16 ; 2 = resid + RES*(x+bias) -> f32
template <int EPI>
__global__ void k_gemm_bf16(const us* __restrict__ A,
                            const us* __restrict__ Bt,
                            const float* __restrict__ bias,
                            const float* __restrict__ resid,
                            void* __restrict__ Cout,
                            int Ndim, int Kdim) {
    constexpr int BM = 128, BN = 128, BK = 32, LD = 40;  // padded 80B LDS rows
    __shared__ __align__(16) us As[2][BM * LD];
    __shared__ __align__(16) us Bs[2][BN * LD];

    const int tid = threadIdx.x;
    const int bm = blockIdx.y * BM, bn = blockIdx.x * BN;
    const int wid = tid >> 5, lane = tid & 31;
    const int r = lane & 15, hi = lane >> 4;
    const int wm0 = (wid >> 1) * 32;   // 4 waves along M
    const int wn0 = (wid & 1) * 64;    // 2 waves along N

    // tile-copy mapping: half an LDS row (16 elements, 32 contiguous bytes) per thread
    const int grow = tid >> 1;               // 0..127
    const int gkc  = (tid & 1) * 16;         // 0 or 16 (elements)
    const us* gA = &A [(size_t)(bm + grow) * Kdim + gkc];
    const us* gB = &Bt[(size_t)(bn + grow) * Kdim + gkc];
    const int lo0 = grow * LD + gkc;

    v8f acc[2][4];
#pragma unroll
    for (int i = 0; i < 2; ++i)
#pragma unroll
        for (int j = 0; j < 4; ++j) acc[i][j] = (v8f)0.0f;

    const int KT = Kdim / BK;
    // prologue: async-copy tile 0 into buffer 0 (global -> LDS, no VGPRs)
    async_ld_b128(&As[0][lo0],     gA);
    async_ld_b128(&As[0][lo0 + 8], gA + 8);
    async_ld_b128(&Bs[0][lo0],     gB);
    async_ld_b128(&Bs[0][lo0 + 8], gB + 8);

    int buf = 0;
    for (int kt = 0; kt < KT; ++kt) {
        wait_async0();          // our async chunks have landed in LDS
        __syncthreads();        // everyone's chunks have landed
        if (kt + 1 < KT) {      // async-copy next tile into the other buffer
            const int k1 = (kt + 1) * BK;
            const int nb = buf ^ 1;
            async_ld_b128(&As[nb][lo0],     gA + k1);
            async_ld_b128(&As[nb][lo0 + 8], gA + k1 + 8);
            async_ld_b128(&Bs[nb][lo0],     gB + k1);
            async_ld_b128(&Bs[nb][lo0 + 8], gB + k1 + 8);
            if (kt + 2 < KT) {  // global_prefetch_b8 two tiles ahead
                __builtin_prefetch(gA + (kt + 2) * BK, 0, 1);
                __builtin_prefetch(gB + (kt + 2) * BK, 0, 1);
            }
        }

        v16bf af[2], bfr[4];
#pragma unroll
        for (int i = 0; i < 2; ++i) af[i]  = frag_ld(&As[buf][(wm0 + 16 * i + r) * LD], hi);
#pragma unroll
        for (int j = 0; j < 4; ++j) bfr[j] = frag_ld(&Bs[buf][(wn0 + 16 * j + r) * LD], hi);
#pragma unroll
        for (int i = 0; i < 2; ++i)
#pragma unroll
            for (int j = 0; j < 4; ++j) acc[i][j] = wmma_bf16(af[i], bfr[j], acc[i][j]);
        buf ^= 1;
    }

#pragma unroll
    for (int i = 0; i < 2; ++i)
#pragma unroll
        for (int j = 0; j < 4; ++j) {
            const int gn = bn + wn0 + 16 * j + r;
            const float bv = bias[gn];
#pragma unroll
            for (int v = 0; v < 8; ++v) {
                const int gm = bm + wm0 + 16 * i + v + 8 * hi;
                const size_t off = (size_t)gm * Ndim + gn;
                float val = acc[i][j][v] + bv;
                if constexpr (EPI == 0) {
                    ((us*)Cout)[off] = f2bf(val);
                } else if constexpr (EPI == 1) {
                    ((us*)Cout)[off] = f2bf(fmaxf(val, 0.0f));
                } else {
                    ((float*)Cout)[off] = resid[off] + RES_C * val;
                }
            }
        }
}

// -------- causal flash attention, transposed tiles (S^T = K*Q^T) -----------
// q,k,v,o : bf16 [B, S, D] with column = h*HDIM + d
// Double-buffered K/V tiles, ONE barrier per key-block; next tile's K-async
// and V-global loads issue right after the barrier so HBM latency hides
// behind the WMMA/softmax phase.  Softmax state is scalar per lane
// (query = q0 + (lane&15)); reductions are in-lane VALU + v_permlanex16.
// All fragments for each WMMA burst are preloaded so the wmma ops can fire
// back-to-back under a single dscnt wait.
__global__ void k_attention(const us* __restrict__ q,
                            const us* __restrict__ k,
                            const us* __restrict__ v,
                            us* __restrict__ o) {
    __shared__ __align__(16) us Ks[2][32 * 72];    // 32 keys x 64 d (pad 72)
    __shared__ __align__(16) us Vt[2][64 * 40];    // 64 d x 32 keys (pad 40)

    const int b = blockIdx.z, h = blockIdx.y, qb = blockIdx.x * 128;
    const int tid = threadIdx.x;
    const int wid = tid >> 5, lane = tid & 31;
    const int r = lane & 15, hi = lane >> 4;
    const int q0 = qb + wid * 16;
    const int qg = q0 + r;                         // this lane's query row
    const size_t basebh = (size_t)b * SEQ * DMODEL + h * HDIM;

    // Q^T B-fragments (K = d, two 32-wide slices of HDIM=64)
    v16bf qf[2];
#pragma unroll
    for (int t = 0; t < 2; ++t)
        qf[t] = frag_ld(q + basebh + (size_t)(q0 + r) * DMODEL + t * 32, hi);

    v8f o4[4];                                     // O^T: rows d, cols query
#pragma unroll
    for (int j = 0; j < 4; ++j) o4[j] = (v8f)0.0f;
    float m_st = -1e30f, l_st = 0.0f;              // m in log2 units

    const int nkb = qb / 32 + 4;                   // causal: keys <= qb+127

    // K-tile copy slot: 1 async b128 per thread; V-tile: key-pair packing,
    // 2 x b64 global loads (one tile ahead) -> 4 x b32 LDS stores
    const int kkey = tid >> 3, kdc = (tid & 7) * 8;
    const int vkp  = (tid >> 4) * 2;               // even key of the pair
    const int vdc  = (tid & 15) * 4;               // 4 d's per thread
    const us* kptr = k + basebh + (size_t)kkey * DMODEL + kdc;
    const us* vptr = v + basebh + (size_t)vkp * DMODEL + vdc;
    const int kslot = kkey * 72 + kdc;

    // prologue: tile 0 in flight
    async_ld_b128(&Ks[0][kslot], kptr);
    v4u va = *(const v4u*)(vptr);
    v4u vb = *(const v4u*)(vptr + DMODEL);
    kptr += 32 * DMODEL;
    vptr += 32 * DMODEL;

    int buf = 0;
    for (int kb = 0; kb < nkb; ++kb) {
        const int kbase = kb * 32;
        // stage this tile's V regs into Vt[buf]
#pragma unroll
        for (int j = 0; j < 4; ++j) {
            const unsigned comb = (unsigned)va[j] | ((unsigned)vb[j] << 16);
            *(unsigned*)&Vt[buf][(vdc + j) * 40 + vkp] = comb;
        }
        wait_async0();          // our K chunk landed
        __syncthreads();        // everyone's K + V staged

        if (kb + 1 < nkb) {     // issue next tile's loads into the other buffer
            async_ld_b128(&Ks[buf ^ 1][kslot], kptr);
            va = *(const v4u*)(vptr);
            vb = *(const v4u*)(vptr + DMODEL);
            kptr += 32 * DMODEL;
            vptr += 32 * DMODEL;
        }

        if (kbase <= q0 + 15) {                    // wave-uniform causal skip
            // S^T[32k x 16q] = K(32x64) * Q^T : preload all 4 K-fragments,
            // then 4 back-to-back WMMAs
            v16bf kf[2][2];                        // [key-tile][d-step]
#pragma unroll
            for (int t = 0; t < 2; ++t)
#pragma unroll
                for (int d = 0; d < 2; ++d)
                    kf[t][d] = frag_ld(&Ks[buf][(t * 16 + r) * 72 + d * 32], hi);
            v8f st[2];
            st[0] = (v8f)0.0f; st[1] = (v8f)0.0f;
#pragma unroll
            for (int d = 0; d < 2; ++d)
#pragma unroll
                for (int t = 0; t < 2; ++t)
                    st[t] = wmma_bf16(kf[t][d], qf[d], st[t]);

            // log2-domain scores; mask only for diagonal-edge blocks
            float sv[2][8];
            if (kbase + 31 <= q0) {                // fully unmasked (uniform)
#pragma unroll
                for (int t = 0; t < 2; ++t)
#pragma unroll
                    for (int vv = 0; vv < 8; ++vv) sv[t][vv] = st[t][vv] * SCL2_C;
            } else {
#pragma unroll
                for (int t = 0; t < 2; ++t)
#pragma unroll
                    for (int vv = 0; vv < 8; ++vv) {
                        const int key_i = kbase + 16 * t + 8 * hi + vv;
                        const float xs = st[t][vv] * SCL2_C;
                        sv[t][vv] = (key_i > qg) ? -1e30f : xs;
                    }
            }
            // block max for this query: in-lane tree + one half-swap
            float mx = sv[0][0];
#pragma unroll
            for (int t = 0; t < 2; ++t)
#pragma unroll
                for (int vv = 0; vv < 8; ++vv) mx = fmaxf(mx, sv[t][vv]);
            mx = fmaxf(mx, xhalf_swap(mx));
            const float mnew = fmaxf(m_st, mx);
            const float a = __builtin_amdgcn_exp2f(m_st - mnew);

            float ps = 0.0f;
            v8u pk[2];
#pragma unroll
            for (int t = 0; t < 2; ++t)
#pragma unroll
                for (int vv = 0; vv < 8; ++vv) {
                    const float p = __builtin_amdgcn_exp2f(sv[t][vv] - mnew);
                    ps += p;
                    pk[t][vv] = f2bf(p);
                }
            ps += xhalf_swap(ps);
            l_st = l_st * a + ps;
            m_st = mnew;

#pragma unroll
            for (int j = 0; j < 4; ++j)
#pragma unroll
                for (int vv = 0; vv < 8; ++vv) o4[j][vv] *= a;

            // O^T += V^T(64x32) * P^T(32x16); preload all 4 V-fragments,
            // P^T fragment straight from regs, then 4 back-to-back WMMAs
            const v16bf pf = frag_pack(pk[0], pk[1]);
            v16bf vf[4];
#pragma unroll
            for (int j = 0; j < 4; ++j)
                vf[j] = frag_ld(&Vt[buf][(j * 16 + r) * 40], hi);
#pragma unroll
            for (int j = 0; j < 4; ++j)
                o4[j] = wmma_bf16(vf[j], pf, o4[j]);
        }
        buf ^= 1;
    }

    // normalize and store bf16 [B,S,D]; per lane: 4 packed 16B stores
    const float rinv = 1.0f / l_st;
    const size_t obase = ((size_t)b * SEQ + qg) * DMODEL + h * HDIM;
#pragma unroll
    for (int j = 0; j < 4; ++j) {
        v8u pk;
#pragma unroll
        for (int vv = 0; vv < 8; ++vv) pk[vv] = f2bf(o4[j][vv] * rinv);
        *(v8u*)&o[obase + 16 * j + 8 * hi] = pk;
    }
}

// ---------------- host launcher ----------------
extern "C" void kernel_launch(void* const* d_in, const int* in_sizes, int n_in,
                              void* d_out, int out_size, void* d_ws, size_t ws_size,
                              hipStream_t stream) {
    const float* x   = (const float*)d_in[0];
    const float* Wq  = (const float*)d_in[1];
    const float* bq  = (const float*)d_in[2];
    const float* Wk  = (const float*)d_in[3];
    const float* bk  = (const float*)d_in[4];
    const float* Wv  = (const float*)d_in[5];
    const float* bv  = (const float*)d_in[6];
    const float* Wo  = (const float*)d_in[7];
    const float* bo  = (const float*)d_in[8];
    const float* g1  = (const float*)d_in[9];
    const float* be1 = (const float*)d_in[10];
    const float* g2  = (const float*)d_in[11];
    const float* be2 = (const float*)d_in[12];
    const float* W1  = (const float*)d_in[13];
    const float* b1  = (const float*)d_in[14];
    const float* W2  = (const float*)d_in[15];
    const float* b2  = (const float*)d_in[16];
    float* out = (float*)d_out;

    char* base = (char*)d_ws;
    size_t off = 0;
    auto alloc = [&](size_t n) -> void* {
        off = (off + 255) & ~(size_t)255;
        void* r = base + off; off += n; return r;
    };
    us* ln1  = (us*)alloc((size_t)MTOK * DMODEL * 2);
    us* Wqt  = (us*)alloc((size_t)DMODEL * DMODEL * 2);
    us* Wkt  = (us*)alloc((size_t)DMODEL * DMODEL * 2);
    us* Wvt  = (us*)alloc((size_t)DMODEL * DMODEL * 2);
    us* Wot  = (us*)alloc((size_t)DMODEL * DMODEL * 2);
    us* W1t  = (us*)alloc((size_t)DMODEL * DFFN * 2);
    us* W2t  = (us*)alloc((size_t)DMODEL * DFFN * 2);
    us* qbuf = (us*)alloc((size_t)MTOK * DMODEL * 2);
    us* kbuf = (us*)alloc((size_t)MTOK * DMODEL * 2);
    us* vbuf = (us*)alloc((size_t)MTOK * DMODEL * 2);
    us* attn = (us*)alloc((size_t)MTOK * DMODEL * 2);
    float* x1 = (float*)alloc((size_t)MTOK * DMODEL * 4);
    us* ln2  = (us*)alloc((size_t)MTOK * DMODEL * 2);
    us* h1   = (us*)alloc((size_t)MTOK * DFFN * 2);

    k_transpose_bf16<<<dim3(DMODEL / 32, DMODEL / 32), 256, 0, stream>>>(Wq, Wqt, DMODEL, DMODEL);
    k_transpose_bf16<<<dim3(DMODEL / 32, DMODEL / 32), 256, 0, stream>>>(Wk, Wkt, DMODEL, DMODEL);
    k_transpose_bf16<<<dim3(DMODEL / 32, DMODEL / 32), 256, 0, stream>>>(Wv, Wvt, DMODEL, DMODEL);
    k_transpose_bf16<<<dim3(DMODEL / 32, DMODEL / 32), 256, 0, stream>>>(Wo, Wot, DMODEL, DMODEL);
    k_transpose_bf16<<<dim3(DFFN / 32,   DMODEL / 32), 256, 0, stream>>>(W1, W1t, DMODEL, DFFN);
    k_transpose_bf16<<<dim3(DMODEL / 32, DFFN / 32),   256, 0, stream>>>(W2, W2t, DFFN, DMODEL);

    k_layernorm<<<MTOK, 256, 0, stream>>>(x, g1, be1, ln1);

    dim3 gD(DMODEL / 128, MTOK / 128);
    dim3 gF(DFFN / 128,  MTOK / 128);
    k_gemm_bf16<0><<<gD, 256, 0, stream>>>(ln1, Wqt, bq, nullptr, qbuf, DMODEL, DMODEL);
    k_gemm_bf16<0><<<gD, 256, 0, stream>>>(ln1, Wkt, bk, nullptr, kbuf, DMODEL, DMODEL);
    k_gemm_bf16<0><<<gD, 256, 0, stream>>>(ln1, Wvt, bv, nullptr, vbuf, DMODEL, DMODEL);

    k_attention<<<dim3(SEQ / 128, NHEAD, BDIM), 256, 0, stream>>>(qbuf, kbuf, vbuf, attn);

    k_gemm_bf16<2><<<gD, 256, 0, stream>>>(attn, Wot, bo, x, x1, DMODEL, DMODEL);

    k_layernorm<<<MTOK, 256, 0, stream>>>(x1, g2, be2, ln2);

    k_gemm_bf16<1><<<gF, 256, 0, stream>>>(ln2, W1t, b1, nullptr, h1, DFFN, DMODEL);
    k_gemm_bf16<2><<<gD, 256, 0, stream>>>(h1, W2t, b2, x1, out, DMODEL, DFFN);
}
